// LSTMLayer_18305150616456
// MI455X (gfx1250) — matrix operand order repaired and checked by reading, not compiled, not run on hardware
//
#include <hip/hip_runtime.h>
#include <hip/hip_bf16.h>

// ---------------------------------------------------------------------------
// LSTM layer for MI455X (gfx1250, wave32, WMMA).
// T=512, B=64, I=H=1024.  bf16 WMMA 16x16x32, f32 accumulate.
//
// Per timestep kernel: grid 64 blocks (one per 16-wide hidden-column tile),
// 128 threads (4 waves = 4 batch tiles of 16).  Fused K-loop over 2048
// (x-projection K=0..1023, recurrent K=1024..2047), LDS double-buffered A
// operand staged with async global->LDS when available.
// ---------------------------------------------------------------------------

typedef __attribute__((ext_vector_type(16))) __bf16 v16bf;
typedef __attribute__((ext_vector_type(8)))  float  v8f;
typedef int v4i_t __attribute__((vector_size(16)));   // builtin's b128 unit
typedef __attribute__((address_space(1))) v4i_t* gptr_b128;
typedef __attribute__((address_space(3))) v4i_t* lptr_b128;

#define T_STEPS 512
#define BATCH   64
#define HID     1024
#define KCHUNKS 64      // 2048 / 32

#if defined(__has_builtin)
#  if __has_builtin(__builtin_amdgcn_global_load_async_to_lds_b128)
#    define HAVE_ASYNC_B128 1
#  endif
#  if __has_builtin(__builtin_amdgcn_s_wait_asynccnt)
#    define HAVE_WAIT_ASYNC 1
#  endif
#endif
#ifndef HAVE_ASYNC_B128
#  define HAVE_ASYNC_B128 0
#endif
#ifndef HAVE_WAIT_ASYNC
#  define HAVE_WAIT_ASYNC 0
#endif

// Position of original k-index j (0..31) inside the stored 32-wide K chunk.
// Storage order [k0..7, k16..23, k8..15, k24..31] so that lanes 0-15 of a
// wave read their (K=0..7,16..23) half as one contiguous 32B chunk and lanes
// 16-31 read (K=8..15,24..31) as the next 32B (per ISA 16-bit A layout).
__host__ __device__ __forceinline__ int kperm(int j) {
  if (j < 8)  return j;        // 0..7   -> 0..7
  if (j < 16) return j + 8;    // 8..15  -> 16..23
  if (j < 24) return j - 8;    // 16..23 -> 8..15
  return j;                    // 24..31 -> 24..31
}

__device__ __forceinline__ float sigm_f(float x) {
  return 1.0f / (1.0f + __expf(-x));
}
__device__ __forceinline__ float tanh_f(float x) {
  float e = __expf(-2.0f * fabsf(x));
  float t = (1.0f - e) / (1.0f + e);
  return copysignf(t, x);
}

__device__ __forceinline__ void wait_async_lds() {
#if HAVE_ASYNC_B128
#  if HAVE_WAIT_ASYNC
  __builtin_amdgcn_s_wait_asynccnt(0);
#  else
  asm volatile("s_wait_asynccnt 0" ::: "memory");
#  endif
#endif
}

// ---------------------------------------------------------------------------
// Prep kernel 1: pack W_ih (K=0..1023) and W_hh (K=1024..2047) into bf16
// WMMA-B tiles: layout [ntile(256)][kchunk(64)][512 elems], where within a
// 32x16 tile, lane l holds column N=l&15, K=kperm((l>>4)*16 + 0..15).
// ---------------------------------------------------------------------------
__global__ void pack_weights(const float* __restrict__ w_ih,
                             const float* __restrict__ w_hh,
                             __bf16* __restrict__ wpack) {
  const int e     = blockIdx.x * 256 + threadIdx.x;   // 0 .. 8,388,607
  const int ntile = e >> 15;                          // 0..255
  const int rem   = e & 32767;
  const int ks    = rem >> 9;                         // 0..63
  const int idx   = rem & 511;
  const int l     = idx >> 4;                         // lane 0..31
  const int kk    = idx & 15;
  const int gcol  = ntile * 16 + (l & 15);            // row of W (0..4095)
  const int ksrc  = ks * 32 + kperm((l >> 4) * 16 + kk); // 0..2047
  const float v = (ksrc < 1024) ? w_ih[(size_t)gcol * 1024 + ksrc]
                                : w_hh[(size_t)gcol * 1024 + (ksrc - 1024)];
  wpack[e] = (__bf16)v;
}

// Prep kernel 2: input f32 -> bf16 with per-32-chunk K permutation.
__global__ void pack_input(const float* __restrict__ x,
                           __bf16* __restrict__ xbf) {
  const size_t e   = (size_t)blockIdx.x * 256 + threadIdx.x; // 0..33,554,431
  const size_t row = e >> 10;
  const int    col = (int)(e & 1023);
  xbf[row * 1024 + (size_t)(col & ~31) + kperm(col & 31)] = (__bf16)x[e];
}

// Prep kernel 3: h0 -> permuted bf16 state buffer, c0 -> f32 cell buffer.
__global__ void init_state(const float* __restrict__ h0,
                           const float* __restrict__ c0,
                           __bf16* __restrict__ hbf0,
                           float* __restrict__ cbuf) {
  const int e   = blockIdx.x * 256 + threadIdx.x;  // 0..65535
  const int col = e & 1023;
  hbf0[(e & ~1023) + (col & ~31) + kperm(col & 31)] = (__bf16)h0[e];
  cbuf[e] = c0[e];
}

// ---------------------------------------------------------------------------
// Per-timestep fused GEMM + gate kernel.
// ---------------------------------------------------------------------------
__global__ __launch_bounds__(128)
void lstm_step(const __bf16* __restrict__ xbf_t,    // [64][1024] permuted bf16
               const __bf16* __restrict__ hbf_in,   // [64][1024] permuted bf16
               __bf16* __restrict__ hbf_out,        // next-step A operand
               const __bf16* __restrict__ wpack,    // packed B tiles
               const float* __restrict__ b_ih,
               const float* __restrict__ b_hh,
               float* __restrict__ cbuf,            // [64][1024] cell state
               float* __restrict__ out_t,           // d_out + t*B*H
               float* __restrict__ hf,              // final h (last step) or null
               float* __restrict__ cf) {            // final c (last step) or null
  __shared__ alignas(64) __bf16 abuf[2][BATCH * 32];  // 2 x 4KB A chunks

  const int tid  = threadIdx.x;
  const int lane = tid & 31;
  const int wave = tid >> 5;       // batch tile 0..3
  const int jt   = blockIdx.x;     // hidden-column tile 0..63
  const int ln   = lane & 15;
  const int lh   = lane >> 4;

  // staging role: each thread copies 32 bytes (half an A row) per K chunk
  const int srow  = tid >> 1;      // 0..63
  const int shalf = tid & 1;

  const int j = jt * 16 + ln;      // this lane's hidden column

  // accumulators = 4 gate tiles (i,f,g,o), seeded with the fused bias.
  v8f acc[4];
#pragma unroll
  for (int g = 0; g < 4; ++g) {
    const float bias = b_ih[g * HID + j] + b_hh[g * HID + j];
#pragma unroll
    for (int v = 0; v < 8; ++v) acc[g][v] = bias;
  }

  auto stage = [&](int ks, int buf) {
    const __bf16* gsrc = (ks < 32)
        ? (xbf_t  + (size_t)srow * 1024 + ks * 32 + shalf * 16)
        : (hbf_in + (size_t)srow * 1024 + (ks - 32) * 32 + shalf * 16);
    __bf16* ldst = &abuf[buf][srow * 32 + shalf * 16];
#if HAVE_ASYNC_B128
    // Builtin signature (from clang): (v4i as1*, v4i as3*, imm off, imm cpol).
    // INST_OFFSET is added to both the global and LDS addresses.
    __builtin_amdgcn_global_load_async_to_lds_b128(
        (gptr_b128)(void*)gsrc, (lptr_b128)(void*)ldst, 0, 0);
    __builtin_amdgcn_global_load_async_to_lds_b128(
        (gptr_b128)(void*)gsrc, (lptr_b128)(void*)ldst, 16, 0);
#else
    *(v16bf*)ldst = *(const v16bf*)gsrc;
#endif
  };

  stage(0, 0);
  wait_async_lds();
  __syncthreads();

  for (int ks = 0; ks < KCHUNKS; ++ks) {
    const int cur = ks & 1;
    if (ks + 1 < KCHUNKS) stage(ks + 1, cur ^ 1);

    // A: 16x32 bf16 tile for this wave's batch rows (ISA 16-bit A layout).
    const v16bf A =
        *(const v16bf*)&abuf[cur][(wave * 16 + ln) * 32 + lh * 16];

#pragma unroll
    for (int g = 0; g < 4; ++g) {
      // B: 32x16 bf16 weight tile for gate g, column tile jt, K chunk ks.
      const v16bf Bm = *(const v16bf*)(
          wpack + ((size_t)((g * 64 + jt) * 64 + ks)) * 512 + lane * 16);
      acc[g] = __builtin_amdgcn_wmma_f32_16x16x32_bf16(
          false, A, false, Bm, (short)0, acc[g], false, false);
    }

    wait_async_lds();
    __syncthreads();
  }

  // Elementwise gate math.  C/D layout: VGPR v, lanes 0-15 -> M=v,
  // lanes 16-31 -> M=v+8; N = lane&15 (constant per lane).
  const int kc  = j >> 5;
  const int pos = kperm(j & 31);
#pragma unroll
  for (int v = 0; v < 8; ++v) {
    const int m = wave * 16 + lh * 8 + v;           // batch row
    const size_t o = (size_t)m * HID + j;
    const float cp = cbuf[o];
    const float ig = sigm_f(acc[0][v]);
    const float fg = sigm_f(acc[1][v]);
    const float gg = tanh_f(acc[2][v]);
    const float og = sigm_f(acc[3][v]);
    const float cn = fg * cp + ig * gg;
    const float hn = og * tanh_f(cn);
    cbuf[o]  = cn;
    out_t[o] = hn;
    hbf_out[(size_t)m * 1024 + kc * 32 + pos] = (__bf16)hn;
    if (hf) { hf[o] = hn; cf[o] = cn; }
  }
}

// ---------------------------------------------------------------------------
// Host launcher.  Workspace layout (~80.5 MB):
//   [0, 16MB)        wpack  : packed bf16 weights (L2-resident working set)
//   [16MB, 80MB)     xbf    : bf16 permuted input, all timesteps
//   [80MB, +128KB)   hbf0   : bf16 permuted h state (ping)
//   [.., +128KB)     hbf1   : bf16 permuted h state (pong)
//   [.., +256KB)     cbuf   : f32 cell state
// ---------------------------------------------------------------------------
extern "C" void kernel_launch(void* const* d_in, const int* in_sizes, int n_in,
                              void* d_out, int out_size, void* d_ws,
                              size_t ws_size, hipStream_t stream) {
  (void)in_sizes; (void)n_in; (void)out_size; (void)ws_size;

  const float* input = (const float*)d_in[0];
  const float* h0    = (const float*)d_in[1];
  const float* c0    = (const float*)d_in[2];
  const float* w_ih  = (const float*)d_in[3];
  const float* w_hh  = (const float*)d_in[4];
  const float* b_ih  = (const float*)d_in[5];
  const float* b_hh  = (const float*)d_in[6];

  char* ws = (char*)d_ws;
  __bf16* wpack = (__bf16*)(ws);
  __bf16* xbf   = (__bf16*)(ws + (16ull << 20));
  __bf16* hbf0  = (__bf16*)(ws + (80ull << 20));
  __bf16* hbf1  = hbf0 + (size_t)BATCH * HID;
  float*  cbuf  = (float*)(ws + (80ull << 20) +
                           2ull * BATCH * HID * sizeof(__bf16));

  pack_weights<<<32768, 256, 0, stream>>>(w_ih, w_hh, wpack);
  pack_input  <<<131072, 256, 0, stream>>>(input, xbf);
  init_state  <<<256, 256, 0, stream>>>(h0, c0, hbf0, cbuf);

  float* out = (float*)d_out;
  float* hf  = out + (size_t)T_STEPS * BATCH * HID;
  float* cf  = hf + (size_t)BATCH * HID;

  for (int t = 0; t < T_STEPS; ++t) {
    const __bf16* hin  = (t & 1) ? hbf1 : hbf0;
    __bf16*       hout = (t & 1) ? hbf0 : hbf1;
    const bool last = (t == T_STEPS - 1);
    lstm_step<<<64, 128, 0, stream>>>(
        xbf + (size_t)t * BATCH * 1024, hin, hout, wpack, b_ih, b_hh, cbuf,
        out + (size_t)t * BATCH * HID, last ? hf : nullptr,
        last ? cf : nullptr);
  }
}
